// DynamicBottleneck_55559696941738
// MI455X (gfx1250) — matrix operand order repaired and checked by reading, not compiled, not run on hardware
//
#include <hip/hip_runtime.h>
#include <hip/hip_bf16.h>

// ---------------------------------------------------------------------------
// Types for WMMA
// ---------------------------------------------------------------------------
typedef __attribute__((ext_vector_type(16))) __bf16 v16bf;
typedef __attribute__((ext_vector_type(8)))  float  v8f;
typedef __attribute__((ext_vector_type(4)))  unsigned int u32x4;

union Frag16 {
    u32x4 q[2];   // 32 bytes
    v16bf v;      // 16 bf16 = one WMMA 16-bit operand fragment per lane
};

union H2 {        // two bf16 packed in a dword
    __bf16   h[2];
    unsigned u;
};

// ---------------------------------------------------------------------------
// Problem constants
// ---------------------------------------------------------------------------
#define CIN   256
#define HH    100
#define WWID  100
#define HWSZ  10000
#define BATCH 8
#define NPIX  80000            // BATCH * HWSZ
#define KTAP  9                // 3x3
#define CCHUNKS 8              // 256 / 32
#define KCHUNK 32

#define BM 64                  // Cout tile per block
#define BN 64                  // pixel tile per block
#define BKPD 20                // LDS row pitch in dwords (32 bf16 + pad, conflict-free)

#define NFRAG (KTAP*CCHUNKS*16)        // 1152 A-fragments per conv (16 co-tiles)
#define FRAG_ELEMS 512                 // 32 lanes * 16 bf16

#define GN_EPS 1e-5f
#define TAU    0.2f

// ---------------------------------------------------------------------------
// Repack conv weights [Co][Ci][3][3] fp32 into WMMA A-fragment layout,
// split into hi/lo bf16 (x = hi + lo) for fp32-like accuracy.
// A 16x32 bf16 layout (per ISA 7.12.2):
//   lane<16 : M=lane,    v0..3 -> K=2v,2v+1 ; v4..7 -> K=16+2(v-4),+1
//   lane>=16: M=lane-16, K += 8 (v0..3 -> 8..15 ; v4..7 -> 24..31)
// ---------------------------------------------------------------------------
__global__ void repack_w(const float* __restrict__ W,
                         __bf16* __restrict__ Ahi, __bf16* __restrict__ Alo)
{
    int t = blockIdx.x * blockDim.x + threadIdx.x;   // frag*32 + lane
    if (t >= NFRAG * 32) return;
    int lane  = t & 31;
    int frag  = t >> 5;          // chunk*16 + co_tile
    int tile  = frag & 15;
    int chunk = frag >> 4;       // r*8 + cc
    int cc    = chunk & 7;
    int r     = chunk >> 3;
    int m     = lane & 15;
    int co    = tile * 16 + m;
    int kadd  = (lane >= 16) ? 8 : 0;
    size_t ob = (size_t)frag * FRAG_ELEMS + (size_t)lane * 16;

#pragma unroll
    for (int v = 0; v < 8; ++v) {
        int k0 = ((v < 4) ? 2 * v : 16 + 2 * (v - 4)) + kadd;
#pragma unroll
        for (int e = 0; e < 2; ++e) {
            int kk = k0 + e;
            int ci = cc * KCHUNK + kk;
            float val = W[(size_t)co * (CIN * 9) + (size_t)ci * 9 + r];
            __bf16 h = (__bf16)val;
            __bf16 l = (__bf16)(val - (float)h);
            Ahi[ob + v * 2 + e] = h;
            Alo[ob + v * 2 + e] = l;
        }
    }
}

// ---------------------------------------------------------------------------
// Implicit-GEMM 3x3 conv (stride 1, pad 1):  Y[b,co,h,w] = sum W*X
// Block: 256 threads (8 waves), tile 64(co) x 64(pix).
// Wave grid 2(M) x 4(N): each wave owns 32 co x 16 pix = two 16x16 f32 accs.
// Split-bf16: acc += Ah*Bh + Ah*Bl + Al*Bh   (lo*lo dropped, ~1e-4 rel err)
// B tile double-buffered in LDS -> one barrier per K-chunk; staging stores
// packed bf16 pairs as dwords (halves DS traffic, uses v_cvt_pk_bf16_f32).
// ---------------------------------------------------------------------------
__global__ __launch_bounds__(256)
void conv3x3_wmma(const float* __restrict__ X,
                  const __bf16* __restrict__ Ahi, const __bf16* __restrict__ Alo,
                  float* __restrict__ Y)
{
    __shared__ unsigned Bs_hi[2][BN * BKPD];
    __shared__ unsigned Bs_lo[2][BN * BKPD];

    const int tid  = threadIdx.x;
    const int lane = tid & 31;
    const int wave = tid >> 5;
    const int wm   = wave >> 2;      // 0..1  (M)
    const int wn   = wave & 3;       // 0..3  (N)

    const int bn   = blockIdx.x;     // 1250 pixel blocks
    const int bm   = blockIdx.y;     // 4 co blocks
    const int pix0 = bn * BN;

    // B-tile staging mapping: thread -> (n, dword-column)
    const int ln = tid & 63;         // pixel within tile
    const int lk = tid >> 6;         // 0..3 ; dword col kp = pass*4 + lk
    const int p    = pix0 + ln;
    const int pb   = p / HWSZ;
    const int prem = p - pb * HWSZ;
    const int ph   = prem / WWID;
    const int pw   = prem - ph * WWID;

    v8f acc0 = {};
    v8f acc1 = {};

    const int nn   = lane & 15;
    const int kbdw = (lane >> 4) << 2;            // B frag K base in dwords: 0 or 4
    const int brow = wn * 16 + nn;                // pixel row within B tile

    for (int r = 0; r < KTAP; ++r) {
        const int dh = r / 3 - 1;
        const int dw = r % 3 - 1;
        const int yy = ph + dh;
        const int xx = pw + dw;
        const bool inb = (yy >= 0) && (yy < HH) && (xx >= 0) && (xx < WWID);
        const size_t base = inb
            ? ((size_t)pb * CIN * HWSZ + (size_t)yy * WWID + (size_t)xx) : 0;

        for (int cc = 0; cc < CCHUNKS; ++cc) {
            const int c0  = cc * KCHUNK;
            const int buf = cc & 1;               // (r*8+cc)&1 == cc&1
            // -------- stage B tile (packed bf16 pairs) --------
#pragma unroll
            for (int pass = 0; pass < 4; ++pass) {
                int kp  = pass * 4 + lk;          // dword column 0..15
                int kk  = kp * 2;
                float v0 = 0.f, v1 = 0.f;
                if (inb) {
                    v0 = X[base + (size_t)(c0 + kk) * HWSZ];
                    v1 = X[base + (size_t)(c0 + kk + 1) * HWSZ];
                }
                H2 hu, lu;
                hu.h[0] = (__bf16)v0;
                hu.h[1] = (__bf16)v1;
                lu.h[0] = (__bf16)(v0 - (float)hu.h[0]);
                lu.h[1] = (__bf16)(v1 - (float)hu.h[1]);
                Bs_hi[buf][ln * BKPD + kp] = hu.u;
                Bs_lo[buf][ln * BKPD + kp] = lu.u;
            }
            __syncthreads();                      // single barrier per chunk

            // -------- A fragments: prepacked, contiguous 32B per lane --------
            const int chunk = r * CCHUNKS + cc;
            const int t0    = bm * 4 + wm * 2;    // first of two co-tiles
            Frag16 a0h, a0l, a1h, a1l, bh, bl;
            {
                const __bf16* pA;
                pA = Ahi + ((size_t)(chunk * 16 + t0) * FRAG_ELEMS + lane * 16);
                a0h.q[0] = ((const u32x4*)pA)[0]; a0h.q[1] = ((const u32x4*)pA)[1];
                pA = Alo + ((size_t)(chunk * 16 + t0) * FRAG_ELEMS + lane * 16);
                a0l.q[0] = ((const u32x4*)pA)[0]; a0l.q[1] = ((const u32x4*)pA)[1];
                pA = Ahi + ((size_t)(chunk * 16 + t0 + 1) * FRAG_ELEMS + lane * 16);
                a1h.q[0] = ((const u32x4*)pA)[0]; a1h.q[1] = ((const u32x4*)pA)[1];
                pA = Alo + ((size_t)(chunk * 16 + t0 + 1) * FRAG_ELEMS + lane * 16);
                a1l.q[0] = ((const u32x4*)pA)[0]; a1l.q[1] = ((const u32x4*)pA)[1];
            }
            // -------- B fragments from LDS (16B-aligned b128 reads) --------
            {
                const u32x4* pB = (const u32x4*)&Bs_hi[buf][brow * BKPD + kbdw];
                bh.q[0] = pB[0];
                bh.q[1] = pB[2];
                pB = (const u32x4*)&Bs_lo[buf][brow * BKPD + kbdw];
                bl.q[0] = pB[0];
                bl.q[1] = pB[2];
            }

            acc0 = __builtin_amdgcn_wmma_f32_16x16x32_bf16(false, a0h.v, false, bh.v, (short)0, acc0, false, false);
            acc0 = __builtin_amdgcn_wmma_f32_16x16x32_bf16(false, a0h.v, false, bl.v, (short)0, acc0, false, false);
            acc0 = __builtin_amdgcn_wmma_f32_16x16x32_bf16(false, a0l.v, false, bh.v, (short)0, acc0, false, false);
            acc1 = __builtin_amdgcn_wmma_f32_16x16x32_bf16(false, a1h.v, false, bh.v, (short)0, acc1, false, false);
            acc1 = __builtin_amdgcn_wmma_f32_16x16x32_bf16(false, a1h.v, false, bl.v, (short)0, acc1, false, false);
            acc1 = __builtin_amdgcn_wmma_f32_16x16x32_bf16(false, a1l.v, false, bh.v, (short)0, acc1, false, false);
        }
    }

    // Writeback. C/D layout: VGPR j -> M = j + 8*(lane>=16), N = lane&15.
    const int mofs = (lane >> 4) << 3;
    const int pixw = pix0 + wn * 16 + nn;
    const int obat = pixw / HWSZ;
    const int ohw  = pixw - obat * HWSZ;
    float* yb = Y + (size_t)obat * CIN * HWSZ + ohw;
#pragma unroll
    for (int v = 0; v < 8; ++v) {
        int m0 = bm * BM + wm * 32 + mofs + v;
        yb[(size_t)m0 * HWSZ]        = acc0[v];
        yb[(size_t)(m0 + 16) * HWSZ] = acc1[v];
    }
}

// ---------------------------------------------------------------------------
// GroupNorm statistics: one block per (batch, group); 8 channels * 10000 pix
// are contiguous (80000 floats). Writes mean & rstd.
// ---------------------------------------------------------------------------
__global__ void gn_stats(const float* __restrict__ T,
                         float* __restrict__ mean, float* __restrict__ rstd)
{
    const int bgrp = blockIdx.x;                 // 0..255
    const int b = bgrp >> 5, g = bgrp & 31;
    const float4* base =
        (const float4*)(T + (size_t)b * CIN * HWSZ + (size_t)g * 8 * HWSZ);
    float s = 0.f, s2 = 0.f;
    for (int i = threadIdx.x; i < 20000; i += 256) {
        float4 v = base[i];
        s  += v.x + v.y + v.z + v.w;
        s2 += v.x * v.x + v.y * v.y + v.z * v.z + v.w * v.w;
    }
    __shared__ float sh[256], sh2[256];
    sh[threadIdx.x] = s; sh2[threadIdx.x] = s2;
    __syncthreads();
    for (int st = 128; st > 0; st >>= 1) {
        if (threadIdx.x < st) {
            sh[threadIdx.x]  += sh[threadIdx.x + st];
            sh2[threadIdx.x] += sh2[threadIdx.x + st];
        }
        __syncthreads();
    }
    if (threadIdx.x == 0) {
        float mu  = sh[0] * (1.f / 80000.f);
        float var = sh2[0] * (1.f / 80000.f) - mu * mu;
        mean[bgrp] = mu;
        rstd[bgrp] = rsqrtf(fmaxf(var, 0.f) + GN_EPS);
    }
}

// ---------------------------------------------------------------------------
// GN + ReLU apply (after conv1), float4 vectorized.
// ---------------------------------------------------------------------------
__global__ void gn_apply_relu(const float* __restrict__ T,
                              const float* __restrict__ mean, const float* __restrict__ rstd,
                              const float* __restrict__ gw, const float* __restrict__ gb,
                              float* __restrict__ O)
{
    size_t i = (size_t)blockIdx.x * blockDim.x + threadIdx.x;   // float4 index
    if (i >= 5120000) return;
    size_t e = i * 4;
    int c = (int)((e / HWSZ) & 255);
    int b = (int)(e / (CIN * HWSZ));
    int bgrp = b * 32 + (c >> 3);
    float rs = rstd[bgrp];
    float w  = gw[c] * rs;
    float bb = gb[c] - mean[bgrp] * w;
    float4 v = ((const float4*)T)[i];
    v.x = fmaxf(v.x * w + bb, 0.f);
    v.y = fmaxf(v.y * w + bb, 0.f);
    v.z = fmaxf(v.z * w + bb, 0.f);
    v.w = fmaxf(v.w * w + bb, 0.f);
    ((float4*)O)[i] = v;
}

// ---------------------------------------------------------------------------
// SpatialGate: g = relu(tanh(dot(x[b,:,hw], Wg) + bg - tau)) per pixel.
// ---------------------------------------------------------------------------
__global__ void gate_kernel(const float* __restrict__ X,
                            const float* __restrict__ Wg, const float* __restrict__ bg,
                            float* __restrict__ G)
{
    __shared__ float wsh[CIN];
    wsh[threadIdx.x] = Wg[threadIdx.x];
    __syncthreads();
    int p = blockIdx.x * 256 + threadIdx.x;
    if (p >= NPIX) return;
    int b  = p / HWSZ;
    int hw = p - b * HWSZ;
    const float* xb = X + (size_t)b * CIN * HWSZ + hw;
    float s = 0.f;
#pragma unroll 4
    for (int c = 0; c < CIN; ++c) s += xb[(size_t)c * HWSZ] * wsh[c];
    float g = tanhf(s + bg[0] - TAU);
    G[p] = fmaxf(g, 0.f);
}

// ---------------------------------------------------------------------------
// Epilogue: out = relu(GN2(conv2) * g + x), float4 vectorized.
// ---------------------------------------------------------------------------
__global__ void final_kernel(const float* __restrict__ T,
                             const float* __restrict__ mean, const float* __restrict__ rstd,
                             const float* __restrict__ gw, const float* __restrict__ gb,
                             const float* __restrict__ G,
                             const float* __restrict__ X,
                             float* __restrict__ O)
{
    size_t i = (size_t)blockIdx.x * blockDim.x + threadIdx.x;   // float4 index
    if (i >= 5120000) return;
    size_t e = i * 4;
    int c  = (int)((e / HWSZ) & 255);
    int b  = (int)(e / (CIN * HWSZ));
    int hw = (int)(e % HWSZ);
    int bgrp = b * 32 + (c >> 3);
    float rs = rstd[bgrp];
    float w  = gw[c] * rs;
    float bb = gb[c] - mean[bgrp] * w;
    float4 t = ((const float4*)T)[i];
    float4 x = ((const float4*)X)[i];
    float4 g = *(const float4*)(G + (size_t)b * HWSZ + hw);
    float4 o;
    o.x = fmaxf((t.x * w + bb) * g.x + x.x, 0.f);
    o.y = fmaxf((t.y * w + bb) * g.y + x.y, 0.f);
    o.z = fmaxf((t.z * w + bb) * g.z + x.z, 0.f);
    o.w = fmaxf((t.w * w + bb) * g.w + x.w, 0.f);
    ((float4*)O)[i] = o;
}

// ---------------------------------------------------------------------------
// Launch
// ---------------------------------------------------------------------------
extern "C" void kernel_launch(void* const* d_in, const int* in_sizes, int n_in,
                              void* d_out, int out_size, void* d_ws, size_t ws_size,
                              hipStream_t stream)
{
    const float* x     = (const float*)d_in[0];
    const float* W1    = (const float*)d_in[1];
    const float* gn1_w = (const float*)d_in[2];
    const float* gn1_b = (const float*)d_in[3];
    const float* W2    = (const float*)d_in[4];
    const float* gn2_w = (const float*)d_in[5];
    const float* gn2_b = (const float*)d_in[6];
    const float* Wg    = (const float*)d_in[7];
    const float* bg    = (const float*)d_in[8];
    float* out = (float*)d_out;        // g: [0,80000), y: [80000, 20560000)

    char* ws = (char*)d_ws;
    float*  tmpA = (float*)ws;                          // 81,920,000 B
    float*  tmpB = (float*)(ws + 81920000ULL);          // 81,920,000 B
    __bf16* a1h  = (__bf16*)(ws + 163840000ULL);        // 1,179,648 B each
    __bf16* a1l  = a1h + NFRAG * FRAG_ELEMS;
    __bf16* a2h  = a1l + NFRAG * FRAG_ELEMS;
    __bf16* a2l  = a2h + NFRAG * FRAG_ELEMS;
    float*  st   = (float*)(a2l + NFRAG * FRAG_ELEMS);  // 1024 floats
    float* mean1 = st,       *rstd1 = st + 256;
    float* mean2 = st + 512, *rstd2 = st + 768;

    repack_w<<<144, 256, 0, stream>>>(W1, a1h, a1l);
    repack_w<<<144, 256, 0, stream>>>(W2, a2h, a2l);

    conv3x3_wmma<<<dim3(1250, 4), 256, 0, stream>>>(x, a1h, a1l, tmpA);
    gn_stats<<<256, 256, 0, stream>>>(tmpA, mean1, rstd1);
    gn_apply_relu<<<20000, 256, 0, stream>>>(tmpA, mean1, rstd1, gn1_w, gn1_b, tmpB);

    conv3x3_wmma<<<dim3(1250, 4), 256, 0, stream>>>(tmpB, a2h, a2l, tmpA);
    gn_stats<<<256, 256, 0, stream>>>(tmpA, mean2, rstd2);

    gate_kernel<<<313, 256, 0, stream>>>(x, Wg, bg, out);
    final_kernel<<<20000, 256, 0, stream>>>(tmpA, mean2, rstd2, gn2_w, gn2_b,
                                            out, x, out + 80000);
}
